// DeepFM_38577396253263
// MI455X (gfx1250) — matrix operand order repaired and checked by reading, not compiled, not run on hardware
//
#include <hip/hip_runtime.h>
#include <hip/hip_bf16.h>

typedef __attribute__((ext_vector_type(16))) _Float16 v16h;
typedef __attribute__((ext_vector_type(8)))  float    v8f;

#define F_FEAT 39
#define K_EMB  64
#define DIN    (F_FEAT * K_EMB)   // 2496
#define NCHUNK (DIN / 32)         // 78 K-chunks of 32
#define H0 10
#define H1 5
#define H2 3
#define ROWS_PER_WAVE   16
#define WAVES_PER_BLOCK 4
#define BLOCK_THREADS   (WAVES_PER_BLOCK * 32)               // 128
#define ROWS_PER_BLOCK  (ROWS_PER_WAVE * WAVES_PER_BLOCK)    // 64

// B-matrix LDS staging: per chunk, only the 20 lanes with col n < 10 hold real
// data (cols 10..15 of the padded 16-wide output are zero). 16 halfs per slot.
// One extra zeroed slot at the end feeds lanes with n >= 10 (branch-free).
#define BSLOTS 20
#define BCHUNK_HALFS (BSLOTS * 16)            // 320 halfs = 640 B per chunk
#define ZERO_HALF    (NCHUNK * BCHUNK_HALFS)  // 24960 (offset of zero slot)
#define LDSB_HALFS   (ZERO_HALF + 16)         // 24976 halfs = 49,952 B

__global__ __launch_bounds__(BLOCK_THREADS) void deepfm_fused(
    const int*   __restrict__ feat,
    const float* __restrict__ vtab,
    const float* __restrict__ wtab,
    const float* __restrict__ w0s,
    const float* __restrict__ W0,
    const float* __restrict__ b0,
    const float* __restrict__ W1,
    const float* __restrict__ b1,
    const float* __restrict__ W2,
    const float* __restrict__ b2,
    const float* __restrict__ W3,
    const float* __restrict__ b3,
    float*       __restrict__ out)
{
    __shared__ __align__(32) _Float16 ldsB[LDSB_HALFS];     // 49,952 B
    __shared__ float ldsC[WAVES_PER_BLOCK * 256];           //  4,096 B

    const int tid = threadIdx.x;

    // ---- Stage W0 (2496x10 f32) into LDS as f16 WMMA B-matrix tiles ----
    for (int i = tid; i < LDSB_HALFS; i += BLOCK_THREADS) {
        int c    = i / BCHUNK_HALFS;
        int r    = i - c * BCHUNK_HALFS;
        int slot = r >> 4;
        int h    = r & 15;
        int nn   = (slot < 10) ? slot : (slot - 10);
        int kloc = (h < 8) ? (((slot < 10) ? 0 : 8) + h) : (((slot < 10) ? 16 : 24) + (h - 8));
        int gk   = c * 32 + kloc;
        float v  = (i < ZERO_HALF) ? W0[(size_t)gk * H0 + nn] : 0.0f;
        ldsB[i] = (_Float16)v;
    }
    __syncthreads();

    const int wave    = tid >> 5;
    const int lane    = tid & 31;
    const int m       = lane & 15;            // batch row within tile
    const int rowBase = blockIdx.x * ROWS_PER_BLOCK + wave * ROWS_PER_WAVE;
    const int row     = rowBase + m;
    const int off     = (lane < 16) ? 0 : 8;  // A-matrix K-half ownership
    // Branch-free B-operand base (per-lane constant part); -1 => zero slot
    const int bslot16 = (m < H0) ? ((((lane >> 4) * 10) + m) * 16) : -1;

    // FM state: this lane owns 32 of the 64 K positions of its row
    float s[32];
    #pragma unroll
    for (int j = 0; j < 32; ++j) s[j] = 0.0f;
    float qp[4] = {0.0f, 0.0f, 0.0f, 0.0f};   // q partials (break add chain)
    float lin = 0.0f;
    v8f acc0 = {};   // C accumulator, even chunks
    v8f acc1 = {};   // C accumulator, odd chunks

    const int* frow = feat + (size_t)row * F_FEAT;

    // Issue one feature's 32 lane-floats (this lane's A-layout slice)
    auto issue = [&](float4* buf, int idx_) {
        const float* p = vtab + (size_t)idx_ * K_EMB + off;
        buf[0] = *(const float4*)(p + 0);
        buf[1] = *(const float4*)(p + 4);
        buf[2] = *(const float4*)(p + 16);
        buf[3] = *(const float4*)(p + 20);
        buf[4] = *(const float4*)(p + 32);
        buf[5] = *(const float4*)(p + 36);
        buf[6] = *(const float4*)(p + 48);
        buf[7] = *(const float4*)(p + 52);
    };
    // Consume one feature: FM stats + two 16x16x32 WMMAs (split accumulators)
    auto consume = [&](const float4* buf, int f_) {
        // issue both B-operand LDS loads up front
        int ha0 = (bslot16 >= 0) ? ((2 * f_ + 0) * BCHUNK_HALFS + bslot16)
                                 : ZERO_HALF;
        int ha1 = (bslot16 >= 0) ? ((2 * f_ + 1) * BCHUNK_HALFS + bslot16)
                                 : ZERO_HALF;
        const v16h bv0 = *(const v16h*)(ldsB + ha0);
        const v16h bv1 = *(const v16h*)(ldsB + ha1);

        v16h a0, a1;
        #pragma unroll
        for (int ch = 0; ch < 2; ++ch) {
            float4 f0 = buf[ch * 4 + 0];
            float4 f1 = buf[ch * 4 + 1];
            float4 f2 = buf[ch * 4 + 2];
            float4 f3 = buf[ch * 4 + 3];
            float t[16] = { f0.x, f0.y, f0.z, f0.w,  f1.x, f1.y, f1.z, f1.w,
                            f2.x, f2.y, f2.z, f2.w,  f3.x, f3.y, f3.z, f3.w };
            #pragma unroll
            for (int j = 0; j < 16; ++j) {
                s[ch * 16 + j] += t[j];
                qp[j & 3] += t[j] * t[j];
                if (ch == 0) a0[j] = (_Float16)t[j];
                else         a1[j] = (_Float16)t[j];
            }
        }
        acc0 = __builtin_amdgcn_wmma_f32_16x16x32_f16(
                   false, a0, false, bv0, (short)0, acc0, false, false);
        acc1 = __builtin_amdgcn_wmma_f32_16x16x32_f16(
                   false, a1, false, bv1, (short)0, acc1, false, false);
    };

    // ---- software-pipelined gather: manual 2x unroll, two buffers ----
    float4 bufA[8], bufB[8];
    int ia  = frow[0];
    int ib  = frow[1];
    float wa = wtab[ia];           // first-order values pipelined one pair ahead
    float wb = wtab[ib];
    issue(bufA, ia);
    issue(bufB, ib);
    int ia2 = frow[2];
    int ib2 = frow[3];

    #pragma clang loop unroll(disable)
    for (int f = 0; f < F_FEAT - 2; f += 2) {    // pairs (0,1)..(36,37)
        // prefetch indices two pairs ahead
        int ia4 = (f + 4 < F_FEAT) ? frow[f + 4] : ia2;
        int ib4 = (f + 5 < F_FEAT) ? frow[f + 5] : ib2;
        // prefetch first-order values for the NEXT pair (consumed next iter)
        float wa2 = wtab[ia2];
        float wb2 = wtab[ib2];

        consume(bufA, f);
        issue(bufA, ia2);          // loads for feature f+2

        consume(bufB, f + 1);
        issue(bufB, ib2);          // loads for feature f+3 (stale idx ok @end)

        // consume PREVIOUS pair's wtab values (loads are a full iter old)
        lin += wa + wb;
        wa = wa2; wb = wb2;
        ia = ia2; ib = ib2; ia2 = ia4; ib2 = ib4;
    }
    // epilogue: last (odd) feature f = 38, data already in bufA
    lin += wa;
    consume(bufA, F_FEAT - 1);

    v8f acc = acc0 + acc1;

    // ---- bias + ReLU, park h0 tile in per-wave LDS region ----
    // C layout: lanes 0-15 N=lane M=j; lanes 16-31 N=lane-16 M=8+j.
    {
        float bias = (m < H0) ? b0[m] : 0.0f;
        float* cw = ldsC + wave * 256;
        #pragma unroll
        for (int j = 0; j < 8; ++j) {
            int mr = (lane < 16) ? j : (8 + j);
            cw[mr * 16 + m] = fmaxf(acc[j] + bias, 0.0f);
        }
    }
    __syncthreads();

    // ---- FM: combine the two half-K lanes of each row ----
    float sp[4] = {0.0f, 0.0f, 0.0f, 0.0f};
    #pragma unroll
    for (int j = 0; j < 32; ++j) sp[j & 3] += s[j] * s[j];
    float ss = (sp[0] + sp[1]) + (sp[2] + sp[3]);
    float q  = (qp[0] + qp[1]) + (qp[2] + qp[3]);
    float tpart = ss - q;                 // partial ||s||^2 - partial sum V^2
    tpart += __shfl_xor(tpart, 16, 32);
    float fm = 0.5f * tpart;

    // ---- tiny MLP tail (10 -> 5 -> 3 -> 1) on lanes 0..15 ----
    if (lane < 16) {
        const float* cw = ldsC + wave * 256;
        float h0v[H0];
        #pragma unroll
        for (int i = 0; i < H0; ++i) h0v[i] = cw[m * 16 + i];

        float h1v[H1];
        #pragma unroll
        for (int j = 0; j < H1; ++j) {
            float t = b1[j];
            #pragma unroll
            for (int i = 0; i < H0; ++i) t += h0v[i] * W1[i * H1 + j];
            h1v[j] = fmaxf(t, 0.0f);
        }
        float h2v[H2];
        #pragma unroll
        for (int j = 0; j < H2; ++j) {
            float t = b2[j];
            #pragma unroll
            for (int i = 0; i < H1; ++i) t += h1v[i] * W2[i * H2 + j];
            h2v[j] = fmaxf(t, 0.0f);
        }
        float dnn = b3[0];
        #pragma unroll
        for (int i = 0; i < H2; ++i) dnn += h2v[i] * W3[i];

        out[row] = fm + lin + w0s[0] + dnn;
    }
}

extern "C" void kernel_launch(void* const* d_in, const int* in_sizes, int n_in,
                              void* d_out, int out_size, void* d_ws, size_t ws_size,
                              hipStream_t stream) {
    const int*   feat = (const int*)  d_in[0];
    const float* vtab = (const float*)d_in[1];
    const float* wtab = (const float*)d_in[2];
    const float* w0s  = (const float*)d_in[3];
    const float* W0   = (const float*)d_in[4];
    const float* b0   = (const float*)d_in[5];
    const float* W1   = (const float*)d_in[6];
    const float* b1   = (const float*)d_in[7];
    const float* W2   = (const float*)d_in[8];
    const float* b2   = (const float*)d_in[9];
    const float* W3   = (const float*)d_in[10];
    const float* b3   = (const float*)d_in[11];
    float* outp = (float*)d_out;

    const int B = in_sizes[0] / F_FEAT;          // 16384
    const int blocks = B / ROWS_PER_BLOCK;       // 256
    deepfm_fused<<<blocks, BLOCK_THREADS, 0, stream>>>(
        feat, vtab, wtab, w0s, W0, b0, W1, b1, W2, b2, W3, b3, outp);
}
